// BitLinear_56298431316339
// MI455X (gfx1250) — 8-bit matrix operand order repaired and checked by reading, compiled, not run on hardware
//
#include <hip/hip_runtime.h>

typedef __attribute__((ext_vector_type(2))) int v2i;
typedef __attribute__((ext_vector_type(4))) int v4i;
typedef __attribute__((ext_vector_type(8))) int v8i;

#define M_DIM 2048
#define K_DIM 4096
#define N_DIM 4096


__global__ void rms_rows_kernel(const float* __restrict__ x,
                                const float* __restrict__ g,
                                float* __restrict__ rowRms,
                                float* __restrict__ rowAmax) {
  __shared__ float red[256];
  __shared__ float rms_sh;
  const int row = blockIdx.x;
  const int tid = threadIdx.x;
  const float4* xr4 = (const float4*)(x + (size_t)row * K_DIM);
  const float4* g4  = (const float4*)g;

  float s = 0.f;
  #pragma unroll
  for (int i = 0; i < K_DIM / 4 / 256; ++i) {
    float4 v = xr4[tid + i * 256];
    s += v.x * v.x + v.y * v.y + v.z * v.z + v.w * v.w;
  }
  red[tid] = s; __syncthreads();
  for (int off = 128; off > 0; off >>= 1) {
    if (tid < off) red[tid] += red[tid + off];
    __syncthreads();
  }
  if (tid == 0) {
    float rms = rsqrtf(red[0] / (float)K_DIM + 1e-6f);
    rms_sh = rms;
    rowRms[row] = rms;
  }
  __syncthreads();
  const float rms = rms_sh;

  float m = 0.f;
  #pragma unroll
  for (int i = 0; i < K_DIM / 4 / 256; ++i) {
    float4 v = xr4[tid + i * 256];
    float4 w = g4[tid + i * 256];
    m = fmaxf(m, fabsf(v.x * rms * w.x));
    m = fmaxf(m, fabsf(v.y * rms * w.y));
    m = fmaxf(m, fabsf(v.z * rms * w.z));
    m = fmaxf(m, fabsf(v.w * rms * w.w));
  }
  red[tid] = m; __syncthreads();
  for (int off = 128; off > 0; off >>= 1) {
    if (tid < off) red[tid] = fmaxf(red[tid], red[tid + off]);
    __syncthreads();
  }
  if (tid == 0) rowAmax[row] = red[0];
}

__global__ void ascale_kernel(const float* __restrict__ rowAmax,
                              float* __restrict__ scalars) {
  __shared__ float red[256];
  const int tid = threadIdx.x;
  float m = 0.f;
  for (int i = tid; i < M_DIM; i += 256) m = fmaxf(m, rowAmax[i]);
  red[tid] = m; __syncthreads();
  for (int off = 128; off > 0; off >>= 1) {
    if (tid < off) red[tid] = fmaxf(red[tid], red[tid + off]);
    __syncthreads();
  }
  if (tid == 0) scalars[0] = 127.0f / fmaxf(red[0], 1e-5f);
}

__device__ __forceinline__ int quant_pack4(float4 v, float4 w, float rms, float as) {
  float q0 = fminf(fmaxf(rintf(v.x * rms * w.x * as), -128.f), 127.f);
  float q1 = fminf(fmaxf(rintf(v.y * rms * w.y * as), -128.f), 127.f);
  float q2 = fminf(fmaxf(rintf(v.z * rms * w.z * as), -128.f), 127.f);
  float q3 = fminf(fmaxf(rintf(v.w * rms * w.w * as), -128.f), 127.f);
  return ((int)q0 & 0xff) | (((int)q1 & 0xff) << 8) |
         (((int)q2 & 0xff) << 16) | (((int)q3 & 0xff) << 24);
}

__global__ void aquant_kernel(const float* __restrict__ x,
                              const float* __restrict__ g,
                              const float* __restrict__ rowRms,
                              const float* __restrict__ scalars,
                              signed char* __restrict__ Aq) {
  const int row = blockIdx.x;
  const int tid = threadIdx.x;
  const float rms = rowRms[row];
  const float as  = scalars[0];
  const float4* xr4 = (const float4*)(x + (size_t)row * K_DIM);
  const float4* g4  = (const float4*)g;
  int* ar = (int*)(Aq + (size_t)row * K_DIM);
  #pragma unroll
  for (int i = 0; i < K_DIM / 4 / 256; ++i) {
    int idx = tid + i * 256;
    ar[idx] = quant_pack4(xr4[idx], g4[idx], rms, as);
  }
}

__global__ void wsum_kernel(const float* __restrict__ w,
                            float* __restrict__ part) {
  __shared__ float red[256];
  const int b = blockIdx.x, tid = threadIdx.x;
  const float4* wp = (const float4*)(w + (size_t)b * 4096);
  float s = 0.f;
  #pragma unroll
  for (int i = 0; i < 4; ++i) {
    float4 v = wp[tid + i * 256];
    s += fabsf(v.x) + fabsf(v.y) + fabsf(v.z) + fabsf(v.w);
  }
  red[tid] = s; __syncthreads();
  for (int off = 128; off > 0; off >>= 1) {
    if (tid < off) red[tid] += red[tid + off];
    __syncthreads();
  }
  if (tid == 0) part[b] = red[0];
}

__global__ void bscale_kernel(const float* __restrict__ part,
                              float* __restrict__ scalars) {
  __shared__ float red[256];
  const int tid = threadIdx.x;
  float s = 0.f;
  for (int i = tid; i < 4096; i += 256) s += part[i];
  red[tid] = s; __syncthreads();
  for (int off = 128; off > 0; off >>= 1) {
    if (tid < off) red[tid] += red[tid + off];
    __syncthreads();
  }
  if (tid == 0) {
    float mean = red[0] / (float)((size_t)K_DIM * (size_t)N_DIM);
    scalars[1] = 1.0f / fmaxf(mean, 1e-5f);
  }
}

__global__ void wquant_kernel(const float* __restrict__ w,
                              const float* __restrict__ scalars,
                              signed char* __restrict__ Bq) {
  __shared__ signed char tile[64][68];
  const float bs = scalars[1];
  const int k0 = blockIdx.x * 64;
  const int n0 = blockIdx.y * 64;
  const int tid = threadIdx.x;
  #pragma unroll
  for (int i = 0; i < 4; ++i) {
    int idx = tid + i * 256;
    int r = idx >> 4, c4 = idx & 15;
    const float4 v = ((const float4*)(w + (size_t)(k0 + r) * N_DIM + n0))[c4];
    float q0 = fminf(fmaxf(rintf(v.x * bs), -1.f), 1.f);
    float q1 = fminf(fmaxf(rintf(v.y * bs), -1.f), 1.f);
    float q2 = fminf(fmaxf(rintf(v.z * bs), -1.f), 1.f);
    float q3 = fminf(fmaxf(rintf(v.w * bs), -1.f), 1.f);
    tile[r][c4 * 4 + 0] = (signed char)(int)q0;
    tile[r][c4 * 4 + 1] = (signed char)(int)q1;
    tile[r][c4 * 4 + 2] = (signed char)(int)q2;
    tile[r][c4 * 4 + 3] = (signed char)(int)q3;
  }
  __syncthreads();
  #pragma unroll
  for (int i = 0; i < 4; ++i) {
    int idx = tid + i * 256;
    int r = idx >> 4, c4 = idx & 15;
    int packed = ((int)(unsigned char)tile[c4 * 4 + 0][r]) |
                 ((int)(unsigned char)tile[c4 * 4 + 1][r] << 8) |
                 ((int)(unsigned char)tile[c4 * 4 + 2][r] << 16) |
                 ((int)(unsigned char)tile[c4 * 4 + 3][r] << 24);
    ((int*)(Bq + (size_t)(n0 + r) * K_DIM + k0))[c4] = packed;
  }
}

__device__ __forceinline__ v8i load_a_frag(const signed char* p) {
  v2i t0 = *(const v2i*)(p);
  v2i t1 = *(const v2i*)(p + 16);
  v2i t2 = *(const v2i*)(p + 32);
  v2i t3 = *(const v2i*)(p + 48);
  v8i a;
  a[0] = t0[0]; a[1] = t0[1]; a[2] = t1[0]; a[3] = t1[1];
  a[4] = t2[0]; a[5] = t2[1]; a[6] = t3[0]; a[7] = t3[1];
  return a;
}

__device__ __forceinline__ v8i load_b_frag(const signed char* p) {
  v2i u0 = *(const v2i*)(p);
  v2i u1 = *(const v2i*)(p + 16);
  v2i u2 = *(const v2i*)(p + 32);
  v2i u3 = *(const v2i*)(p + 48);
  v8i b;
  b[0] = u0[0]; b[1] = u0[1]; b[2] = u1[0]; b[3] = u1[1];
  b[4] = u2[0]; b[5] = u2[1]; b[6] = u3[0]; b[7] = u3[1];
  return b;
}

__device__ __forceinline__ v8i wmma_iu8(v8i a, v8i b, v8i c) {
  return __builtin_amdgcn_wmma_i32_16x16x64_iu8(true, a, true, b, c, false, false);
}

__global__ void __launch_bounds__(256)
gemm_iu8_kernel(const signed char* __restrict__ Aq,
                const signed char* __restrict__ Bq,
                const float* __restrict__ scalars,
                float* __restrict__ out) {
  const int lane = threadIdx.x & 31;
  const int wave = threadIdx.x >> 5;
  const int h  = lane >> 4;
  const int rc = lane & 15;
  const int mtile = blockIdx.y * 256 + wave * 32;
  const int ntile = blockIdx.x * 64;

  const signed char* arow0 = Aq + (size_t)(mtile + rc) * K_DIM + h * 8;
  const signed char* arow1 = arow0 + (size_t)16 * K_DIM;
  const signed char* bcol0 = Bq + (size_t)(ntile + rc) * K_DIM + h * 8;
  const signed char* bcol1 = bcol0 + (size_t)16 * K_DIM;
  const signed char* bcol2 = bcol0 + (size_t)32 * K_DIM;
  const signed char* bcol3 = bcol0 + (size_t)48 * K_DIM;

  v8i c00 = {0,0,0,0,0,0,0,0}, c01 = {0,0,0,0,0,0,0,0};
  v8i c02 = {0,0,0,0,0,0,0,0}, c03 = {0,0,0,0,0,0,0,0};
  v8i c10 = {0,0,0,0,0,0,0,0}, c11 = {0,0,0,0,0,0,0,0};
  v8i c12 = {0,0,0,0,0,0,0,0}, c13 = {0,0,0,0,0,0,0,0};

  #pragma unroll 1
  for (int k = 0; k < K_DIM; k += 64) {
    const v8i a0 = load_a_frag(arow0 + k);
    const v8i a1 = load_a_frag(arow1 + k);
    const v8i b0 = load_b_frag(bcol0 + k);
    const v8i b1 = load_b_frag(bcol1 + k);
    const v8i b2 = load_b_frag(bcol2 + k);
    const v8i b3 = load_b_frag(bcol3 + k);

    c00 = wmma_iu8(a0, b0, c00);
    c01 = wmma_iu8(a0, b1, c01);
    c02 = wmma_iu8(a0, b2, c02);
    c03 = wmma_iu8(a0, b3, c03);
    c10 = wmma_iu8(a1, b0, c10);
    c11 = wmma_iu8(a1, b1, c11);
    c12 = wmma_iu8(a1, b2, c12);
    c13 = wmma_iu8(a1, b3, c13);
  }

  const float inv = 1.0f / (scalars[0] * scalars[1]);
  float* orow0 = out + (size_t)(mtile + h * 8) * N_DIM + ntile + rc;
  float* orow1 = orow0 + (size_t)16 * N_DIM;
  #pragma unroll
  for (int i = 0; i < 8; ++i) {
    float* p0 = orow0 + (size_t)i * N_DIM;
    p0[0]  = (float)c00[i] * inv;
    p0[16] = (float)c01[i] * inv;
    p0[32] = (float)c02[i] * inv;
    p0[48] = (float)c03[i] * inv;
    float* p1 = orow1 + (size_t)i * N_DIM;
    p1[0]  = (float)c10[i] * inv;
    p1[16] = (float)c11[i] * inv;
    p1[32] = (float)c12[i] * inv;
    p1[48] = (float)c13[i] * inv;
  }
}

extern "C" void kernel_launch(void* const* d_in, const int* in_sizes, int n_in,
                              void* d_out, int out_size, void* d_ws, size_t ws_size,
                              hipStream_t stream) {
  const float* x = (const float*)d_in[0];
  const float* w = (const float*)d_in[1];
  const float* g = (const float*)d_in[2];
  float* out = (float*)d_out;

  char* ws = (char*)d_ws;
  float* scalars = (float*)(ws);
  float* rowRms  = (float*)(ws + 64);
  float* rowAmax = (float*)(ws + 8256);
  float* wPart   = (float*)(ws + 16448);
  signed char* Aq = (signed char*)(ws + 65536);
  signed char* Bq = (signed char*)(ws + 65536 + (size_t)M_DIM * K_DIM);

  rms_rows_kernel<<<M_DIM, 256, 0, stream>>>(x, g, rowRms, rowAmax);
  ascale_kernel<<<1, 256, 0, stream>>>(rowAmax, scalars);
  aquant_kernel<<<M_DIM, 256, 0, stream>>>(x, g, rowRms, scalars, Aq);

  wsum_kernel<<<4096, 256, 0, stream>>>(w, wPart);
  bscale_kernel<<<1, 256, 0, stream>>>(wPart, scalars);
  dim3 wq_grid(K_DIM / 64, N_DIM / 64);
  wquant_kernel<<<wq_grid, 256, 0, stream>>>(w, scalars, Bq);

  dim3 gemm_grid(N_DIM / 64, M_DIM / 256);
  gemm_iu8_kernel<<<gemm_grid, 256, 0, stream>>>(Aq, Bq, scalars, out);
}
